// Experts_3521873183476
// MI455X (gfx1250) — matrix operand order repaired and checked by reading, not compiled, not run on hardware
//
#include <hip/hip_runtime.h>
#include <type_traits>

// ---------------------------------------------------------------------------
// Grouped-expert FFN for MI455X (gfx1250, wave32, WMMA).
//   h   = gelu(x_e @ w1_e + b1_e)          (bf16 intermediate in d_ws)
//   out = h @ w2_e + b2_e                  (fp32, flat [E][B][N][D] order)
// Both GEMMs use v_wmma_f32_16x16x32_bf16 with f32 accumulation, with a
// register-prefetch pipeline so global loads overlap the WMMA stream.
// ---------------------------------------------------------------------------

typedef __attribute__((ext_vector_type(16))) __bf16 bf16x16;
typedef __attribute__((ext_vector_type(8)))  __bf16 bf16x8;
typedef __attribute__((ext_vector_type(4)))  __bf16 bf16x4;
typedef __attribute__((ext_vector_type(8)))  float  floatx8;

constexpr int kB = 2, kE = 8, kN = 2048, kD = 1024, kH = 4096;
constexpr int kM = kB * kN;          // 4096 rows per expert
constexpr int TILE  = 128;           // 128x128 output tile per workgroup
constexpr int KSTEP = 32;            // bf16 WMMA K
constexpr int LDA   = 40;            // padded LDS stride (breaks bank conflicts)
constexpr int LDB   = 40;

static_assert(kM % TILE == 0 && kH % TILE == 0 && kD % TILE == 0, "tiles");
static_assert(kD % KSTEP == 0 && kH % KSTEP == 0, "ksteps");

// tanh-GELU == x * sigmoid(2*sqrt(2/pi)*(x + 0.044715 x^3)).
// __expf lowers to branch-free native v_exp_f32 (no exec-mask divergence).
__device__ __forceinline__ float gelu_fast(float x) {
    const float k2c0 = 1.5957691216057308f;   // 2*sqrt(2/pi)
    const float c1   = 0.044715f;
    float x3 = x * x * x;
    float t  = -k2c0 * (x + c1 * x3);
    return x / (1.0f + __expf(t));
}

// MODE 1: A = x (fp32, [B][E][N][D]),  W = w1 ([E][D][H]),  Out = h (bf16)
// MODE 2: A = h (bf16, [E][M][H]),     W = w2 ([E][H][D]),  Out = out (fp32)
template <int MODE>
__global__ __launch_bounds__(256) void ffn_wmma_kernel(
    const void*  __restrict__ Ap,
    const float* __restrict__ Wp,
    const float* __restrict__ biasp,
    void*        __restrict__ Op)
{
    constexpr int KDIM = (MODE == 1) ? kD : kH;
    constexpr int NDIM = (MODE == 1) ? kH : kD;
    using ARegT = std::conditional_t<MODE == 1, float4, bf16x4>;

    const int e     = blockIdx.z;
    const int tileN = blockIdx.x * TILE;
    const int tileM = blockIdx.y * TILE;

    __shared__ __bf16 Ash[TILE * LDA];   // [row m][k]      (bf16)
    __shared__ __bf16 Bsh[TILE * LDB];   // [col n][k]      (bf16, B-frag major)

    const int t    = threadIdx.x;
    const int wave = t >> 5;
    const int lane = t & 31;
    const int wy   = wave >> 1;          // 4 waves along M (32 rows each)
    const int wx   = wave & 1;           // 2 waves along N (64 cols each)
    const int lhalf = lane >> 4;
    const int lcol  = lane & 15;

    const float* W = Wp + (size_t)e * KDIM * NDIM;

    floatx8 acc[2][4];
    #pragma unroll
    for (int r = 0; r < 2; ++r)
        #pragma unroll
        for (int c = 0; c < 4; ++c)
            acc[r][c] = (floatx8){0.f,0.f,0.f,0.f,0.f,0.f,0.f,0.f};

    // loader thread mapping
    const int arow = t >> 3;             // A: 8 threads/row * 4 elems = 32 k
    const int akq  = (t & 7) * 4;
    const int bkk  = t >> 5;             // B: one k-row per wave, 32 thr * 4 n
    const int bnq  = (t & 31) * 4;

    // Per-thread A source base (row fixed across K-steps).
    // MODE1: x row for grouped row m = b*N+tok lives at ((b*E+e)*N+tok)*D.
    // MODE2: h rows are contiguous at (e*M+m)*H.
    size_t abase[4];
    #pragma unroll
    for (int p = 0; p < 4; ++p) {
        const int m = tileM + arow + p * 32;
        if constexpr (MODE == 1) {
            const int b = m >> 11, tok = m & (kN - 1);
            abase[p] = (((size_t)b * kE + e) * kN + tok) * (size_t)kD + akq;
        } else {
            abase[p] = ((size_t)e * kM + m) * (size_t)kH + akq;
        }
    }

    ARegT  areg[4];
    float4 breg[4];

    auto prefetch = [&](int k0) {
        #pragma unroll
        for (int p = 0; p < 4; ++p) {
            if constexpr (MODE == 1)
                areg[p] = *(const float4*)((const float*)Ap + abase[p] + k0);
            else
                areg[p] = *(const bf16x4*)((const __bf16*)Ap + abase[p] + k0);
        }
        #pragma unroll
        for (int p = 0; p < 4; ++p) {
            const int kr = bkk + p * 8;
            breg[p] = *(const float4*)(W + (size_t)(k0 + kr) * NDIM + tileN + bnq);
        }
    };

    auto commit = [&]() {
        #pragma unroll
        for (int p = 0; p < 4; ++p) {
            bf16x4 v;
            if constexpr (MODE == 1) {
                v.x = (__bf16)areg[p].x; v.y = (__bf16)areg[p].y;
                v.z = (__bf16)areg[p].z; v.w = (__bf16)areg[p].w;
            } else {
                v = areg[p];
            }
            *(bf16x4*)&Ash[(arow + p * 32) * LDA + akq] = v;
        }
        #pragma unroll
        for (int p = 0; p < 4; ++p) {
            const int kr = bkk + p * 8;
            Bsh[(bnq + 0) * LDB + kr] = (__bf16)breg[p].x;
            Bsh[(bnq + 1) * LDB + kr] = (__bf16)breg[p].y;
            Bsh[(bnq + 2) * LDB + kr] = (__bf16)breg[p].z;
            Bsh[(bnq + 3) * LDB + kr] = (__bf16)breg[p].w;
        }
    };

    prefetch(0);

    for (int k0 = 0; k0 < KDIM; k0 += KSTEP) {
        commit();                         // regs (step k) -> LDS
        __syncthreads();

        if (k0 + KSTEP < KDIM)
            prefetch(k0 + KSTEP);         // global loads overlap WMMAs below

        // ---- build WMMA fragments per ISA 7.12.2 layouts -----------------
        bf16x16 afrag[2], bfrag[4];
        #pragma unroll
        for (int r = 0; r < 2; ++r) {
            const int am = wy * 32 + r * 16 + lcol;     // A row = lane%16
            bf16x8 lo = *(const bf16x8*)&Ash[am * LDA + lhalf * 8];
            bf16x8 hi = *(const bf16x8*)&Ash[am * LDA + 16 + lhalf * 8];
            afrag[r] = __builtin_shufflevector(lo, hi,
                0,1,2,3,4,5,6,7,8,9,10,11,12,13,14,15);
        }
        #pragma unroll
        for (int c = 0; c < 4; ++c) {
            const int bn = wx * 64 + c * 16 + lcol;     // B col = lane%16
            bf16x8 lo = *(const bf16x8*)&Bsh[bn * LDB + lhalf * 8];
            bf16x8 hi = *(const bf16x8*)&Bsh[bn * LDB + 16 + lhalf * 8];
            bfrag[c] = __builtin_shufflevector(lo, hi,
                0,1,2,3,4,5,6,7,8,9,10,11,12,13,14,15);
        }

        // ---- 8 WMMAs per K-step ------------------------------------------
        #pragma unroll
        for (int r = 0; r < 2; ++r)
            #pragma unroll
            for (int c = 0; c < 4; ++c)
                acc[r][c] = __builtin_amdgcn_wmma_f32_16x16x32_bf16(
                    false, afrag[r], false, bfrag[c],
                    (short)0, acc[r][c], false, false);

        __syncthreads();
    }

    // ---- epilogue: bias (+gelu) and store --------------------------------
    #pragma unroll
    for (int c = 0; c < 4; ++c) {
        const int n  = tileN + wx * 64 + c * 16 + lcol;
        const float bv = biasp[(size_t)e * NDIM + n];
        #pragma unroll
        for (int r = 0; r < 2; ++r) {
            const floatx8 v = acc[r][c];
            #pragma unroll
            for (int i = 0; i < 8; ++i) {
                const int m = tileM + wy * 32 + r * 16 + lhalf * 8 + i;
                const float o = v[i] + bv;
                if constexpr (MODE == 1) {
                    __bf16* hout = (__bf16*)Op;
                    hout[((size_t)e * kM + m) * kH + n] = (__bf16)gelu_fast(o);
                } else {
                    float* out = (float*)Op;
                    // flat [E][B][N][D] == reference's reshape(B,E,N,D)
                    out[((size_t)e * kM + m) * kD + n] = o;
                }
            }
        }
    }
}

extern "C" void kernel_launch(void* const* d_in, const int* in_sizes, int n_in,
                              void* d_out, int out_size, void* d_ws, size_t ws_size,
                              hipStream_t stream) {
    (void)in_sizes; (void)n_in; (void)out_size; (void)ws_size;
    const float* x  = (const float*)d_in[0];
    const float* w1 = (const float*)d_in[1];
    const float* b1 = (const float*)d_in[2];
    const float* w2 = (const float*)d_in[3];
    const float* b2 = (const float*)d_in[4];

    // bf16 intermediate h: E*M*H * 2 bytes = 256 MiB in d_ws
    void* h = d_ws;

    dim3 blk(256);
    dim3 g1(kH / TILE, kM / TILE, kE);   // 32 x 32 x 8
    ffn_wmma_kernel<1><<<g1, blk, 0, stream>>>(x, w1, b1, h);

    dim3 g2(kD / TILE, kM / TILE, kE);   // 8 x 32 x 8
    ffn_wmma_kernel<2><<<g2, blk, 0, stream>>>(h, w2, b2, d_out);
}
